// MultiHeadAttention_78460462563636
// MI455X (gfx1250) — compile-verified, matrix operand order checked by reading, not run on hardware
//
#include <hip/hip_runtime.h>
#include <hip/hip_bf16.h>

typedef __bf16 bf16;
typedef __attribute__((ext_vector_type(16))) __bf16 v16bf;
typedef __attribute__((ext_vector_type(8)))  float  v8f;
typedef __attribute__((ext_vector_type(4)))  unsigned int v4u;
typedef __attribute__((ext_vector_type(8)))  int v8i;
typedef __attribute__((ext_vector_type(4)))  int v4i;

#define EMB   1024
#define HEADS 16
#define HDIM  64
#define BATCH 8
#define SEQ   1024
#define G4    (4*EMB)
#define KCHUNK 256          // W-panel K chunk staged by the TDM (64 x 256 bf16 = 32KB)

// ---------------------------------------------------------------- helpers
__device__ inline v8f wmma_bf16(v16bf a, v16bf b, v8f c) {
  // D = A(16x32 bf16) x B(32x16 bf16) + C(16x16 f32)
  return __builtin_amdgcn_wmma_f32_16x16x32_bf16(false, a, false, b,
                                                 (short)0, c, false, false);
}
__device__ inline v16bf ldfrag(const bf16* p) {
  return *reinterpret_cast<const v16bf*>(p);   // 32B, aligned by construction
}
__device__ inline v8f vzero8() {
  v8f z = {0.f,0.f,0.f,0.f,0.f,0.f,0.f,0.f}; return z;
}
__device__ inline float sigm_f(float x) { return 1.f / (1.f + __expf(-x)); }
__device__ inline float tanh_f(float x) {
  float e = __expf(2.f * x);           // inf/0 safe: 1-2/(e+1) -> +/-1
  return 1.f - 2.f / (e + 1.f);
}

// device-wide generation barrier over blocks sharing counter `bar`.
__device__ inline void gbar(unsigned* bar, unsigned target) {
  __syncthreads();
  if (threadIdx.x == 0) {
    __threadfence();
    __hip_atomic_fetch_add(bar, 1u, __ATOMIC_ACQ_REL, __HIP_MEMORY_SCOPE_AGENT);
    while (__hip_atomic_load(bar, __ATOMIC_ACQUIRE, __HIP_MEMORY_SCOPE_AGENT) < target)
      __builtin_amdgcn_s_sleep(8);
    __threadfence();
  }
  __syncthreads();
}

// Issue a TDM load of a 64-row x KCHUNK-col bf16 panel (row stride K elements)
// from global `gsrc` into LDS offset `ldsoff`. D# per CDNA5 ISA 8.3-8.6:
//  group0: count=1 | lds_addr | global_addr[56:0] | type=2
//  group1: data_size=2B, tensor_dim0=KCHUNK, tensor_dim1=64,
//          tile_dim0=KCHUNK, tile_dim1=64, tensor_dim0_stride=K
//  group2/3 (and trailing group payload): zero (2-D tile)
__device__ inline void tdm_load_panel(const bf16* gsrc, unsigned ldsoff, int K) {
  unsigned long long ga = (unsigned long long)(size_t)gsrc;
  v4u g0 = { 1u,                                   // count=1, user descriptor
             ldsoff,                               // lds_addr (bytes)
             (unsigned)(ga & 0xffffffffull),       // global_addr[31:0]
             (unsigned)((ga >> 32) & 0x01ffffffull) | (2u << 30) };  // [56:32]|type=2
  v8i g1 = { (int)(1u << 16),                      // data_size = 2 bytes
             (int)((unsigned)KCHUNK << 16),        // tensor_dim0[15:0]
             (int)(64u << 16),                     // tensor_dim0 hi=0 | tensor_dim1 lo
             (int)((unsigned)KCHUNK << 16),        // tensor_dim1 hi=0 | tile_dim0
             (int)64u,                             // tile_dim1=64 | tile_dim2=0
             (int)K,                               // tensor_dim0_stride[31:0] (elems)
             0, 0 };                               // stride hi | dim1_stride = 0
  v4i g2 = {0, 0, 0, 0};
  v4i g3 = {0, 0, 0, 0};
  v8i g4 = {0, 0, 0, 0, 0, 0, 0, 0};               // unused extra group (this toolchain)
  __builtin_amdgcn_tensor_load_to_lds(g0, g1, g2, g3, g4, 0);
}

// ---------------------------------------------------------------- kernels
__global__ void k_init(unsigned* cnt) {
  if (threadIdx.x < 2) cnt[threadIdx.x] = 0u;
}

__global__ void k_cvt_bf16(const float* __restrict__ in, bf16* __restrict__ out, int n) {
  int i = blockIdx.x * 256 + threadIdx.x;
  if (i < n) out[i] = (bf16)in[i];
}

// vp f32 [B*S, E] -> vhT bf16 [(b*H+h)*64 + d][S]   (PV B-fragment layout)
__global__ void k_vT(const float* __restrict__ vp, bf16* __restrict__ vhT) {
  int i = blockIdx.x * 256 + threadIdx.x;        // 0 .. B*S*E-1
  int e = i & (EMB - 1);
  int rs = i >> 10;                              // b*S + s
  int b = rs >> 10, s = rs & (SEQ - 1);
  int h = e >> 6,  d = e & 63;
  vhT[(((size_t)(b * HEADS + h)) * HDIM + d) * SEQ + s] = (bf16)vp[i];
}

// C[M,N] = A[M,K] @ W[N,K]^T (+bias). Block = 8 waves, tile 128x64.
// W panel is DMA'd into LDS by the Tensor Data Mover (double-buffered
// KCHUNK slices, TENSORcnt-tracked) so the vector pipe only streams A.
__global__ __launch_bounds__(256) void k_gemm_bf16(
    const bf16* __restrict__ A, const bf16* __restrict__ W,
    const float* __restrict__ bias, float* __restrict__ Cf,
    bf16* __restrict__ Cb, int N, int K) {
  const int lane = threadIdx.x & 31, w = threadIdx.x >> 5;
  const int lo = lane & 15, hi = lane >> 4;
  const int row0 = blockIdx.x * 128 + w * 16;
  const int col0 = blockIdx.y * 64;
  __shared__ __attribute__((aligned(1024))) bf16 Wlds[2][64 * KCHUNK];

  const int nchunk = K / KCHUNK;
  const bf16* wpan = W + (size_t)col0 * K;       // 64-row panel base

  v8f acc[4];
#pragma unroll
  for (int n = 0; n < 4; ++n) acc[n] = vzero8();
  const bf16* ap = A + (size_t)(row0 + lo) * K + hi * 16;

  if (w == 0)
    tdm_load_panel(wpan, (unsigned)(size_t)&Wlds[0][0], K);

  for (int c = 0; c < nchunk; ++c) {
    if (c + 1 < nchunk) {
      if (w == 0) {
        tdm_load_panel(wpan + (size_t)(c + 1) * KCHUNK,
                       (unsigned)(size_t)&Wlds[(c + 1) & 1][0], K);
        __builtin_amdgcn_s_wait_tensorcnt(1);    // chunk c landed in LDS
      }
    } else {
      if (w == 0) __builtin_amdgcn_s_wait_tensorcnt(0);
    }
    __syncthreads();
    const bf16* bp = &Wlds[c & 1][(size_t)lo * KCHUNK + hi * 16];
    const bf16* apc = ap + c * KCHUNK;
    __builtin_prefetch(apc + KCHUNK, 0, 3);      // next A slice -> global_prefetch
#pragma unroll
    for (int kk = 0; kk < KCHUNK / 32; ++kk) {
      v16bf a = ldfrag(apc + kk * 32);
#pragma unroll
      for (int n = 0; n < 4; ++n)
        acc[n] = wmma_bf16(a, ldfrag(bp + n * 16 * KCHUNK + kk * 32), acc[n]);
    }
    __syncthreads();                             // buffer reuse fence
  }
#pragma unroll
  for (int n = 0; n < 4; ++n) {
    int col = col0 + n * 16 + lo;
    float bb = bias ? bias[col] : 0.f;
#pragma unroll
    for (int r = 0; r < 8; ++r) {
      int row = row0 + r + 8 * hi;
      float v = acc[n][r] + bb;
      if (Cf) Cf[(size_t)row * N + col] = v;
      else    Cb[(size_t)row * N + col] = (bf16)v;
    }
  }
}

// Persistent dual-LSTM scan. 64 blocks x 256 thr: blocks [0,32)=q, [32,64)=k.
// Each block owns 32 hidden units; waves w: gate=w>>1, 16-col half=(w&1).
// h kept as ping-pong bf16 [16 padded-batch x 1024] buffers (L2 resident).
__global__ __launch_bounds__(256) void k_lstm(
    const float* __restrict__ xg_q, const float* __restrict__ xg_k,
    const bf16* __restrict__ Whhq, const bf16* __restrict__ Whhk,
    const float* __restrict__ bhhq, const float* __restrict__ bhhk,
    bf16* __restrict__ qhb, bf16* __restrict__ khb,
    bf16* __restrict__ hbufq, bf16* __restrict__ hbufk,
    unsigned* __restrict__ cnt) {
  const int grp = blockIdx.x >> 5;                 // 0 = q-LSTM, 1 = k-LSTM
  const int blk = blockIdx.x & 31;
  const float* xg  = grp ? xg_k  : xg_q;
  const bf16*  Whh = grp ? Whhk  : Whhq;
  const float* bhh = grp ? bhhk  : bhhq;
  bf16* hout = grp ? khb : qhb;
  bf16* hbuf = grp ? hbufk : hbufq;
  unsigned* bar = cnt + grp;
  const int j0 = blk * 32;
  const int tid = threadIdx.x;
  const int lane = tid & 31, w = tid >> 5;
  const int lo = lane & 15, hi = lane >> 4;
  __shared__ float gsm[4][32][8];                  // [gate][unit][batch]

  // zero both ping-pong h buffers for this block's 32 columns (all 16 rows)
  for (int idx = tid; idx < 2 * 16 * 32; idx += 256) {
    int bufi = idx >> 9, m = (idx >> 5) & 15, u = idx & 31;
    hbuf[bufi * 16 * EMB + m * EMB + j0 + u] = (bf16)0.f;
  }
  float cstate = 0.f;                              // cell state per (unit,batch)
  const int uu = tid >> 3, mm = tid & 7;           // epilogue mapping
  gbar(bar, 32u);                                  // zeros device-visible

  const int g = w >> 1;
  const int colbase = g * EMB + j0 + (w & 1) * 16; // gate-row base in Whh / xg col
  const bf16* wp = Whh + (size_t)(colbase + lo) * EMB + hi * 16;

  for (int t = 0; t < SEQ; ++t) {
    const bf16* hb = hbuf + (t & 1) * 16 * EMB;
    const bf16* ap = hb + lo * EMB + hi * 16;
    v8f acc = vzero8();
    for (int k0 = 0; k0 < EMB; k0 += 32)
      acc = wmma_bf16(ldfrag(ap + k0), ldfrag(wp + k0), acc);
    if (hi == 0) {                                 // rows 0..7 = real batches
      int col = colbase + lo;
      float bb = bhh[col];
#pragma unroll
      for (int r = 0; r < 8; ++r)
        gsm[g][(w & 1) * 16 + lo][r] =
            acc[r] + xg[(size_t)(r * SEQ + t) * G4 + col] + bb;
    }
    __syncthreads();
    {
      float iv = sigm_f(gsm[0][uu][mm]);
      float fv = sigm_f(gsm[1][uu][mm]);
      float gv = tanh_f(gsm[2][uu][mm]);
      float ov = sigm_f(gsm[3][uu][mm]);
      cstate = fv * cstate + iv * gv;
      float hv = ov * tanh_f(cstate);
      bf16 h16 = (bf16)hv;
      int j = j0 + uu;
      hbuf[((t + 1) & 1) * 16 * EMB + mm * EMB + j] = h16;   // next-step operand
      int hh = j >> 6, d = j & 63;                           // head layout out
      hout[(((size_t)(mm * HEADS + hh)) * SEQ + t) * HDIM + d] = h16;
    }
    gbar(bar, 32u * (unsigned)(t + 2));
  }
}

// Flash attention, causal. Block = 4 waves = 64 query rows of one (b,head).
__global__ __launch_bounds__(128) void k_flash(
    const bf16* __restrict__ qhb, const bf16* __restrict__ khb,
    const bf16* __restrict__ vhT, const float* __restrict__ mask,
    bf16* __restrict__ concat) {
  const int bh = blockIdx.x >> 4;
  const int q0 = (blockIdx.x & 15) * 64;
  const int b = bh >> 4, head = bh & 15;
  const int lane = threadIdx.x & 31, w = threadIdx.x >> 5;
  const int lo = lane & 15, hi = lane >> 4;
  __shared__ bf16 plds[4][16 * 64];                // per-wave P staging
  const int myq = q0 + w * 16;

  const bf16* qp = qhb + ((size_t)(bh * SEQ + myq + lo)) * HDIM + hi * 16;
  v16bf qf0 = ldfrag(qp);
  v16bf qf1 = ldfrag(qp + 32);

  v8f O[4];
#pragma unroll
  for (int n = 0; n < 4; ++n) O[n] = vzero8();
  float m8[8], l8[8];
#pragma unroll
  for (int r = 0; r < 8; ++r) { m8[r] = -1e30f; l8[r] = 0.f; }

  for (int kt = 0; kt <= q0; kt += 64) {           // causal: skip tiles above diag
    v8f S[4];
#pragma unroll
    for (int n = 0; n < 4; ++n) S[n] = vzero8();
#pragma unroll
    for (int n = 0; n < 4; ++n) {
      const bf16* kp = khb + ((size_t)(bh * SEQ + kt + n * 16 + lo)) * HDIM + hi * 16;
      S[n] = wmma_bf16(qf0, ldfrag(kp), S[n]);
      S[n] = wmma_bf16(qf1, ldfrag(kp + 32), S[n]);
    }
    const bool dm = (kt == q0);                    // diagonal tile: apply real mask
    float rmax[8];
#pragma unroll
    for (int r = 0; r < 8; ++r) rmax[r] = -1e30f;
#pragma unroll
    for (int n = 0; n < 4; ++n)
#pragma unroll
      for (int r = 0; r < 8; ++r) {
        float v = S[n][r] * 0.125f;                // 1/sqrt(64)
        if (dm) v += mask[(size_t)(myq + r + 8 * hi) * SEQ + kt + n * 16 + lo];
        S[n][r] = v;
        rmax[r] = fmaxf(rmax[r], v);
      }
#pragma unroll
    for (int r = 0; r < 8; ++r)
      for (int off = 1; off < 16; off <<= 1)
        rmax[r] = fmaxf(rmax[r], __shfl_xor(rmax[r], off, 32));
    float alpha[8];
#pragma unroll
    for (int r = 0; r < 8; ++r) {
      float mn = fmaxf(m8[r], rmax[r]);
      alpha[r] = __expf(m8[r] - mn);
      m8[r] = mn;
    }
#pragma unroll
    for (int n = 0; n < 4; ++n)
#pragma unroll
      for (int r = 0; r < 8; ++r) S[n][r] = __expf(S[n][r] - m8[r]);
#pragma unroll
    for (int r = 0; r < 8; ++r) {
      float s = S[0][r] + S[1][r] + S[2][r] + S[3][r];
      for (int off = 1; off < 16; off <<= 1) s += __shfl_xor(s, off, 32);
      l8[r] = l8[r] * alpha[r] + s;
    }
#pragma unroll
    for (int n = 0; n < 4; ++n)
#pragma unroll
      for (int r = 0; r < 8; ++r) O[n][r] *= alpha[r];
    // C-layout -> A-fragment re-shape through per-wave LDS
#pragma unroll
    for (int n = 0; n < 4; ++n)
#pragma unroll
      for (int r = 0; r < 8; ++r)
        plds[w][(r + 8 * hi) * 64 + n * 16 + lo] = (bf16)S[n][r];
    v16bf p0 = *reinterpret_cast<const v16bf*>(&plds[w][lo * 64 + hi * 16]);
    v16bf p1 = *reinterpret_cast<const v16bf*>(&plds[w][lo * 64 + 32 + hi * 16]);
#pragma unroll
    for (int n = 0; n < 4; ++n) {
      const bf16* vp = vhT + ((size_t)(bh * HDIM + n * 16 + lo)) * SEQ + kt + hi * 16;
      O[n] = wmma_bf16(p0, ldfrag(vp), O[n]);
      O[n] = wmma_bf16(p1, ldfrag(vp + 32), O[n]);
    }
  }
  float inv[8];
#pragma unroll
  for (int r = 0; r < 8; ++r) inv[r] = 1.f / l8[r];
#pragma unroll
  for (int n = 0; n < 4; ++n)
#pragma unroll
    for (int r = 0; r < 8; ++r)
      concat[(size_t)(b * SEQ + myq + r + 8 * hi) * EMB + head * HDIM + n * 16 + lo] =
          (bf16)(O[n][r] * inv[r]);
}

// ---------------------------------------------------------------- host
extern "C" void kernel_launch(void* const* d_in, const int* in_sizes, int n_in,
                              void* d_out, int out_size, void* d_ws, size_t ws_size,
                              hipStream_t stream) {
  (void)in_sizes; (void)n_in; (void)out_size; (void)ws_size;
  const float* q     = (const float*)d_in[0];
  const float* k     = (const float*)d_in[1];
  const float* v     = (const float*)d_in[2];
  const float* mask  = (const float*)d_in[3];
  const float* Wih_q = (const float*)d_in[4];
  const float* Whh_q = (const float*)d_in[5];
  const float* bih_q = (const float*)d_in[6];
  const float* bhh_q = (const float*)d_in[7];
  const float* Wih_k = (const float*)d_in[8];
  const float* Whh_k = (const float*)d_in[9];
  const float* bih_k = (const float*)d_in[10];
  const float* bhh_k = (const float*)d_in[11];
  const float* Wv    = (const float*)d_in[12];
  const float* Wout  = (const float*)d_in[13];
  float* out = (float*)d_out;

  const size_t M = (size_t)BATCH * SEQ;            // 8192 token rows
  char* base = (char*)d_ws; size_t off = 0;
  auto carve = [&](size_t bytes) -> void* {
    off = (off + 255) & ~(size_t)255;
    void* p = base + off; off += bytes; return p;
  };
  bf16* qb    = (bf16*)carve(M * EMB * 2);
  bf16* kb    = (bf16*)carve(M * EMB * 2);
  bf16* vb    = (bf16*)carve(M * EMB * 2);
  bf16* Wihqb = (bf16*)carve((size_t)G4 * EMB * 2);
  bf16* Wihkb = (bf16*)carve((size_t)G4 * EMB * 2);
  bf16* Whhqb = (bf16*)carve((size_t)G4 * EMB * 2);
  bf16* Whhkb = (bf16*)carve((size_t)G4 * EMB * 2);
  bf16* Wvb   = (bf16*)carve((size_t)EMB * EMB * 2);
  bf16* Woutb = (bf16*)carve((size_t)EMB * EMB * 2);
  float* xgq  = (float*)carve(M * G4 * 4);
  float* xgk  = (float*)carve(M * G4 * 4);
  float* vp   = (float*)carve(M * EMB * 4);
  bf16* qhb   = (bf16*)carve(M * EMB * 2);         // [b,h,s,d]
  bf16* khb   = (bf16*)carve(M * EMB * 2);
  bf16* vhT   = (bf16*)carve(M * EMB * 2);         // [b,h,d,s]
  bf16* conc  = (bf16*)carve(M * EMB * 2);
  bf16* hbq   = (bf16*)carve(2 * 16 * EMB * 2);    // ping-pong h (q)
  bf16* hbk   = (bf16*)carve(2 * 16 * EMB * 2);    // ping-pong h (k)
  unsigned* cnt = (unsigned*)carve(2 * sizeof(unsigned));

  k_init<<<1, 64, 0, stream>>>(cnt);

  const int NQ = (int)(M * EMB), NW4 = G4 * EMB, NW1 = EMB * EMB;
  k_cvt_bf16<<<NQ / 256, 256, 0, stream>>>(q, qb, NQ);
  k_cvt_bf16<<<NQ / 256, 256, 0, stream>>>(k, kb, NQ);
  k_cvt_bf16<<<NQ / 256, 256, 0, stream>>>(v, vb, NQ);
  k_cvt_bf16<<<NW4 / 256, 256, 0, stream>>>(Wih_q, Wihqb, NW4);
  k_cvt_bf16<<<NW4 / 256, 256, 0, stream>>>(Wih_k, Wihkb, NW4);
  k_cvt_bf16<<<NW4 / 256, 256, 0, stream>>>(Whh_q, Whhqb, NW4);
  k_cvt_bf16<<<NW4 / 256, 256, 0, stream>>>(Whh_k, Whhkb, NW4);
  k_cvt_bf16<<<NW1 / 256, 256, 0, stream>>>(Wv, Wvb, NW1);
  k_cvt_bf16<<<NW1 / 256, 256, 0, stream>>>(Wout, Woutb, NW1);

  // input-side gate projections + value projection (TDM-staged W panels)
  k_gemm_bf16<<<dim3(64, 64), 256, 0, stream>>>(qb, Wihqb, bih_q, xgq, nullptr, G4, EMB);
  k_gemm_bf16<<<dim3(64, 64), 256, 0, stream>>>(kb, Wihkb, bih_k, xgk, nullptr, G4, EMB);
  k_gemm_bf16<<<dim3(64, 16), 256, 0, stream>>>(vb, Wvb, nullptr, vp, nullptr, EMB, EMB);

  // both LSTM scans, persistent, device-barrier per step
  k_lstm<<<64, 256, 0, stream>>>(xgq, xgk, Whhqb, Whhkb, bhh_q, bhh_k,
                                 qhb, khb, hbq, hbk, cnt);

  k_vT<<<NQ / 256, 256, 0, stream>>>(vp, vhT);

  // causal flash attention -> concat (bf16, feeds final GEMM)
  k_flash<<<BATCH * HEADS * (SEQ / 64), 128, 0, stream>>>(qhb, khb, vhT, mask, conc);

  // output projection -> f32 d_out
  k_gemm_bf16<<<dim3(64, 16), 256, 0, stream>>>(conc, Woutb, nullptr, out, nullptr, EMB, EMB);
}
